// CQAttention_10934986735936
// MI455X (gfx1250) — matrix operand order repaired and checked by reading, not compiled, not run on hardware
//
#include <hip/hip_runtime.h>
#include <hip/hip_bf16.h>

// ---------------------------------------------------------------------------
// CQAttention for MI455X (gfx1250, wave32, WMMA bf16 16x16x32, f32 accum)
//
// Key optimization: Bm = (S1 @ S2^T) @ Ct is reassociated as
//   T  = S2^T @ Ct            (512x128 per batch,  8.6 GFLOP)
//   Bm = S1 @ T               (same shape as A-GEMM, 8.6 GFLOP)
// cutting total FLOPs from ~103 GF to ~34 GF and eliminating the
// 1024x1024 S12 intermediate entirely.
// ---------------------------------------------------------------------------
#define B_  64
#define D_  128
#define LC_ 1024
#define LQ_ 512
#define NEG_ (-1e30f)

typedef __attribute__((ext_vector_type(16))) __bf16 v16bf;
typedef __attribute__((ext_vector_type(8)))  __bf16 v8bf;
typedef __attribute__((ext_vector_type(8)))  float  v8f;

// ---- WMMA helpers ---------------------------------------------------------

__device__ inline v8f wmma_bf16(v16bf a, v16bf b, v8f c) {
  return __builtin_amdgcn_wmma_f32_16x16x32_bf16(
      /*neg_a=*/false, a, /*neg_b=*/false, b,
      /*c_mod=*/(short)0, c, /*reuse_a=*/false, /*reuse_b=*/false);
}

__device__ inline v8f v8f_zero() {
  v8f z;
#pragma unroll
  for (int r = 0; r < 8; ++r) z[r] = 0.0f;
  return z;
}

// A-matrix (16x32 bf16, M-major memory, row stride ld):
// lane l: m = l&15, half = l>>4.  elements 0..7 -> K = half*8 + e,
//                                 elements 8..15 -> K = half*8 + 16 + (e-8)
__device__ inline v16bf load_a_tile(const __bf16* base, int ld) {
  const int lane = threadIdx.x & 31;
  const int m = lane & 15;
  const int k0 = (lane >> 4) * 8;
  const __bf16* p = base + (size_t)m * ld + k0;
  v8bf lo = *(const v8bf*)(p);
  v8bf hi = *(const v8bf*)(p + 16);
  return __builtin_shufflevector(lo, hi, 0, 1, 2, 3, 4, 5, 6, 7,
                                 8, 9, 10, 11, 12, 13, 14, 15);
}

// B-matrix (32x16 KxN bf16) loaded from an N-major ("BT") buffer, row stride ld:
// lane l: n = l&15, kbase = (l>>4)*8, elements 0..7 -> K = kbase + e, 8..15 -> K = kbase + 16 + (e-8).
__device__ inline v16bf load_bt_tile(const __bf16* base, int ld) {
  const int lane = threadIdx.x & 31;
  const int n = lane & 15;
  const int kb = (lane >> 4) * 8;
  const __bf16* p = base + (size_t)n * ld + kb;
  v8bf lo = *(const v8bf*)(p);
  v8bf hi = *(const v8bf*)(p + 16);
  return __builtin_shufflevector(lo, hi, 0, 1, 2, 3, 4, 5, 6, 7,
                                 8, 9, 10, 11, 12, 13, 14, 15);
}

// ---- Pass 0: s0[b,c] = Ct.w4C, s1[b,q] = Qt.w4Q ---------------------------

__global__ void scalars_kernel(const float* __restrict__ C,
                               const float* __restrict__ Q,
                               const float* __restrict__ w4C,
                               const float* __restrict__ w4Q,
                               float* __restrict__ s0,
                               float* __restrict__ s1) {
  int idx = blockIdx.x * 256 + threadIdx.x;   // grid sized exactly
  int b = idx / (LC_ + LQ_);
  int r = idx % (LC_ + LQ_);
  float acc = 0.0f;
  if (r < LC_) {
#pragma unroll 4
    for (int d = 0; d < D_; ++d)
      acc += C[((size_t)b * D_ + d) * LC_ + r] * w4C[d];
    s0[(size_t)b * LC_ + r] = acc;
  } else {
    int q = r - LC_;
#pragma unroll 4
    for (int d = 0; d < D_; ++d)
      acc += Q[((size_t)b * D_ + d) * LQ_ + q] * w4Q[d];
    s1[(size_t)b * LQ_ + q] = acc;
  }
}

// ---- Pass 1a: straight f32 -> bf16 cast -----------------------------------

__global__ void cast_bf16_kernel(const float* __restrict__ src,
                                 __bf16* __restrict__ dst, size_t n) {
  size_t i = (size_t)blockIdx.x * blockDim.x + threadIdx.x;
  size_t stride = (size_t)gridDim.x * blockDim.x;
  for (; i < n; i += stride) dst[i] = (__bf16)src[i];
}

// ---- Pass 1b: transpose-cast: src (B, D, L) -> dst (B, L, D), optional scale[d]

__global__ void transpose_cast_kernel(const float* __restrict__ src,
                                      const float* __restrict__ scale,
                                      __bf16* __restrict__ dst, int L) {
  __shared__ float tile[32][33];
  const int b = blockIdx.z;
  const int l0 = blockIdx.x * 32;
  const int r0 = blockIdx.y * 32;
  const int tx = threadIdx.x & 31;
  const int ty = threadIdx.x >> 5;  // 0..7
#pragma unroll
  for (int yy = 0; yy < 32; yy += 8) {
    int r = r0 + ty + yy;
    float sc = scale ? scale[r] : 1.0f;
    tile[ty + yy][tx] = src[((size_t)b * D_ + r) * L + l0 + tx] * sc;
  }
  __syncthreads();
#pragma unroll
  for (int yy = 0; yy < 32; yy += 8) {
    dst[((size_t)b * L + l0 + ty + yy) * D_ + r0 + tx] = (__bf16)tile[tx][ty + yy];
  }
}

// ---- Pass 2: S = (Ct*w4mlu) @ Qt^T + s0 + s1 + bias  (WMMA) ---------------
// Each wave: one 16(M) x 64(N) tile, K = D_ = 128.

__global__ void s_gemm_kernel(const __bf16* __restrict__ CtWb,
                              const __bf16* __restrict__ Qtb,
                              const float* __restrict__ s0,
                              const float* __restrict__ s1,
                              const float* __restrict__ bias,
                              float* __restrict__ S) {
  const int wave = threadIdx.x >> 5;
  const int lane = threadIdx.x & 31;
  const int gw = blockIdx.x * 8 + wave;
  const int mTiles = LC_ / 16, nGroups = LQ_ / 64;
  const int b = gw / (mTiles * nGroups);
  const int rem = gw % (mTiles * nGroups);
  const int mt = rem / nGroups;
  const int ng = rem % nGroups;

  const __bf16* Abase = CtWb + ((size_t)b * LC_ + mt * 16) * D_;
  const __bf16* Bbase = Qtb + ((size_t)b * LQ_ + ng * 64) * D_;

  v8f acc[4];
#pragma unroll
  for (int t = 0; t < 4; ++t) acc[t] = v8f_zero();

#pragma unroll
  for (int kt = 0; kt < D_; kt += 32) {
    v16bf a = load_a_tile(Abase + kt, D_);
#pragma unroll
    for (int t = 0; t < 4; ++t) {
      v16bf bm = load_bt_tile(Bbase + (size_t)t * 16 * D_ + kt, D_);
      acc[t] = wmma_bf16(a, bm, acc[t]);
    }
  }

  const int half = lane >> 4, nl = lane & 15;
  const float bv = bias[0];
#pragma unroll
  for (int t = 0; t < 4; ++t) {
    int n = ng * 64 + t * 16 + nl;
    float s1v = s1[(size_t)b * LQ_ + n] + bv;
#pragma unroll
    for (int r = 0; r < 8; ++r) {
      int m = mt * 16 + half * 8 + r;
      S[((size_t)b * LC_ + m) * LQ_ + n] = acc[t][r] + s0[(size_t)b * LC_ + m] + s1v;
    }
  }
}

// ---- Pass 3a: row softmax (axis q) -> S1 bf16 (c-major) -------------------

__global__ void row_softmax_kernel(const float* __restrict__ S,
                                   const float* __restrict__ Qmask,
                                   __bf16* __restrict__ S1b) {
  const int wave = threadIdx.x >> 5;
  const int lane = threadIdx.x & 31;
  const int b = blockIdx.y;
  const int c = blockIdx.x * 8 + wave;
  const float* row = S + ((size_t)b * LC_ + c) * LQ_;
  const float* qm = Qmask + (size_t)b * LQ_;

  float x[16];
  float mx = -3.4e38f;
#pragma unroll
  for (int i = 0; i < 16; ++i) {
    int q = lane + 32 * i;
    float qv = qm[q];
    x[i] = row[q] * qv + (1.0f - qv) * NEG_;
    mx = fmaxf(mx, x[i]);
  }
#pragma unroll
  for (int off = 16; off >= 1; off >>= 1) mx = fmaxf(mx, __shfl_xor(mx, off, 32));
  float sum = 0.0f;
#pragma unroll
  for (int i = 0; i < 16; ++i) { x[i] = __expf(x[i] - mx); sum += x[i]; }
#pragma unroll
  for (int off = 16; off >= 1; off >>= 1) sum += __shfl_xor(sum, off, 32);
  float inv = 1.0f / sum;
  __bf16* out = S1b + ((size_t)b * LC_ + c) * LQ_;
#pragma unroll
  for (int i = 0; i < 16; ++i) out[lane + 32 * i] = (__bf16)(x[i] * inv);
}

// ---- Pass 3b: column softmax (axis c) -> S2^T bf16 ------------------------
// Writes S2t[b, q, c]: exactly the M-major A-layout the T-GEMM needs.
// S reads stay coalesced (lane = q); writes are packed 16B v8bf stores.

__global__ void col_softmax_kernel(const float* __restrict__ S,
                                   const float* __restrict__ Cmask,
                                   __bf16* __restrict__ S2t) {
  const int b = blockIdx.y;
  const int q = blockIdx.x * 64 + threadIdx.x;
  const float* cm = Cmask + (size_t)b * LC_;

  float m = -3.4e38f, s = 0.0f;
  for (int c = 0; c < LC_; ++c) {
    float cv = cm[c];
    float x = S[((size_t)b * LC_ + c) * LQ_ + q] * cv + (1.0f - cv) * NEG_;
    float nm = fmaxf(m, x);
    s = s * __expf(m - nm) + __expf(x - nm);
    m = nm;
  }
  float inv = 1.0f / s;
  __bf16* orow = S2t + ((size_t)b * LQ_ + q) * LC_;
  for (int c0 = 0; c0 < LC_; c0 += 8) {
    v8bf pack;
#pragma unroll
    for (int j = 0; j < 8; ++j) {
      float cv = cm[c0 + j];
      float x = S[((size_t)b * LC_ + c0 + j) * LQ_ + q] * cv + (1.0f - cv) * NEG_;
      pack[j] = (__bf16)(__expf(x - m) * inv);
    }
    *(v8bf*)(orow + c0) = pack;
  }
}

// ---- Pass 4: T = S2^T @ Ct  (WMMA), K = LC_ = 1024 ------------------------
// A = S2t (q-major, k contiguous, ld=LC); BT = Cb native (d, c), ld=LC.
// Output stored transposed as Tt[b, d, q] bf16 via packed 16B stores, which is
// exactly the N-major B-operand layout for the Bm GEMM.

__global__ void t_gemm_kernel(const __bf16* __restrict__ S2t,
                              const __bf16* __restrict__ Cb,
                              __bf16* __restrict__ Tt) {
  const int wave = threadIdx.x >> 5;
  const int lane = threadIdx.x & 31;
  const int gw = blockIdx.x * 8 + wave;
  const int mTiles = LQ_ / 16, nGroups = D_ / 64;  // 32, 2
  const int b = gw / (mTiles * nGroups);
  const int rem = gw % (mTiles * nGroups);
  const int mt = rem / nGroups;
  const int ng = rem % nGroups;

  const __bf16* Abase = S2t + ((size_t)b * LQ_ + mt * 16) * LC_;
  const __bf16* Bbase = Cb + ((size_t)b * D_ + ng * 64) * LC_;

  v8f acc[4];
#pragma unroll
  for (int t = 0; t < 4; ++t) acc[t] = v8f_zero();

  for (int kt = 0; kt < LC_; kt += 32) {
    v16bf a = load_a_tile(Abase + kt, LC_);
#pragma unroll
    for (int t = 0; t < 4; ++t) {
      v16bf bm = load_bt_tile(Bbase + (size_t)t * 16 * LC_ + kt, LC_);
      acc[t] = wmma_bf16(a, bm, acc[t]);
    }
  }

  const int half = lane >> 4, nl = lane & 15;
#pragma unroll
  for (int t = 0; t < 4; ++t) {
    int d = ng * 64 + t * 16 + nl;
    v8bf pack;
#pragma unroll
    for (int r = 0; r < 8; ++r) pack[r] = (__bf16)acc[t][r];
    *(v8bf*)(Tt + ((size_t)b * D_ + d) * LQ_ + mt * 16 + half * 8) = pack;
  }
}

// ---- Pass 5: generic  Out[b,c,d] = S1 @ BT  (WMMA), K = LQ_ ---------------
// Used twice: A = S1 @ Qt (BT = native Q layout) and Bm = S1 @ T (BT = Tt).

__global__ void a_gemm_kernel(const __bf16* __restrict__ S1b,
                              const __bf16* __restrict__ BT,
                              float* __restrict__ Out) {
  const int wave = threadIdx.x >> 5;
  const int lane = threadIdx.x & 31;
  const int gw = blockIdx.x * 8 + wave;
  const int mTiles = LC_ / 16, nGroups = D_ / 64;  // 64, 2
  const int b = gw / (mTiles * nGroups);
  const int rem = gw % (mTiles * nGroups);
  const int mt = rem / nGroups;
  const int ng = rem % nGroups;

  const __bf16* Abase = S1b + ((size_t)b * LC_ + mt * 16) * LQ_;
  const __bf16* Bbase = BT + ((size_t)b * D_ + ng * 64) * LQ_;  // BT(n=d, k=q)

  v8f acc[4];
#pragma unroll
  for (int t = 0; t < 4; ++t) acc[t] = v8f_zero();

  for (int kt = 0; kt < LQ_; kt += 32) {
    v16bf a = load_a_tile(Abase + kt, LQ_);
#pragma unroll
    for (int t = 0; t < 4; ++t) {
      v16bf bm = load_bt_tile(Bbase + (size_t)t * 16 * LQ_ + kt, LQ_);
      acc[t] = wmma_bf16(a, bm, acc[t]);
    }
  }

  const int half = lane >> 4, nl = lane & 15;
#pragma unroll
  for (int t = 0; t < 4; ++t) {
#pragma unroll
    for (int r = 0; r < 8; ++r) {
      int m = mt * 16 + half * 8 + r;
      Out[((size_t)b * LC_ + m) * D_ + ng * 64 + t * 16 + nl] = acc[t][r];
    }
  }
}

// ---- Pass 6: out = swap(concat([Ct, A, Ct*A, Ct*Bm], axis=2)) -------------
// out shape (B, 4D, LC); LDS-transpose A/Bm tiles for coalesced writes.

__global__ void output_kernel(const float* __restrict__ C,
                              const float* __restrict__ A,
                              const float* __restrict__ Bm,
                              float* __restrict__ out) {
  __shared__ float As[32][129];
  __shared__ float Bs[32][129];
  const int b = blockIdx.y;
  const int c0 = blockIdx.x * 32;
  const int lx = threadIdx.x & 31;
  const int ly = threadIdx.x >> 5;  // 0..7

#pragma unroll
  for (int cc = 0; cc < 32; cc += 8) {
    int ci = ly + cc;
    const float* arow = A + ((size_t)b * LC_ + c0 + ci) * D_;
    const float* brow = Bm + ((size_t)b * LC_ + c0 + ci) * D_;
#pragma unroll
    for (int k = 0; k < 4; ++k) {
      As[ci][lx + 32 * k] = arow[lx + 32 * k];
      Bs[ci][lx + 32 * k] = brow[lx + 32 * k];
    }
  }
  __syncthreads();

#pragma unroll
  for (int it = 0; it < 16; ++it) {
    int d = ly + 8 * it;  // 0..127
    float cv = C[((size_t)b * D_ + d) * LC_ + c0 + lx];
    float av = As[lx][d];
    float bv = Bs[lx][d];
    size_t ob = ((size_t)b * 4 * D_) * (size_t)LC_ + c0 + lx;
    out[ob + (size_t)d * LC_] = cv;
    out[ob + (size_t)(D_ + d) * LC_] = av;
    out[ob + (size_t)(2 * D_ + d) * LC_] = cv * av;
    out[ob + (size_t)(3 * D_ + d) * LC_] = cv * bv;
  }
}

// ---------------------------------------------------------------------------

extern "C" void kernel_launch(void* const* d_in, const int* in_sizes, int n_in,
                              void* d_out, int out_size, void* d_ws, size_t ws_size,
                              hipStream_t stream) {
  const float* C     = (const float*)d_in[0];
  const float* Q     = (const float*)d_in[1];
  const float* Cmask = (const float*)d_in[2];
  const float* Qmask = (const float*)d_in[3];
  const float* w4C   = (const float*)d_in[4];
  const float* w4Q   = (const float*)d_in[5];
  const float* w4mlu = (const float*)d_in[6];
  const float* bias  = (const float*)d_in[7];
  float* out = (float*)d_out;

  char* ws = (char*)d_ws;
  const size_t MB = 1024ull * 1024ull;
  float*  S    = (float*)(ws + 0);            // 128 MB f32
  __bf16* S1b  = (__bf16*)(ws + 128 * MB);    // 64 MB bf16 (c, q)
  __bf16* S2t  = (__bf16*)(ws + 192 * MB);    // 64 MB bf16 (q, c)  = S2^T
  __bf16* CtWb = (__bf16*)(ws + 256 * MB);    // 16 MB  (c,d) scaled by w4mlu
  __bf16* Cb   = (__bf16*)(ws + 272 * MB);    // 16 MB  native (d,c)
  __bf16* Qb   = (__bf16*)(ws + 288 * MB);    // 8 MB   native (d,q)
  __bf16* Qtb  = (__bf16*)(ws + 296 * MB);    // 8 MB   (q,d)
  float*  s0   = (float*)(ws + 304 * MB);     // 256 KB
  float*  s1   = (float*)(ws + 304 * MB + 256 * 1024); // 128 KB
  __bf16* Tt   = (__bf16*)(ws + 305 * MB);    // 8 MB bf16 (d, q) = T^T
  float*  A    = (float*)(ws + 0);            // 32 MB, aliases dead S region
  float*  Bm   = (float*)(ws + 32 * MB);      // 32 MB, aliases dead S region

  // Pass 0: bias row/col vectors
  scalars_kernel<<<(B_ * (LC_ + LQ_)) / 256, 256, 0, stream>>>(C, Q, w4C, w4Q, s0, s1);

  // Pass 1: bf16 copies (straight casts + transposed casts)
  cast_bf16_kernel<<<2048, 256, 0, stream>>>(Q, Qb, (size_t)B_ * D_ * LQ_);
  cast_bf16_kernel<<<2048, 256, 0, stream>>>(C, Cb, (size_t)B_ * D_ * LC_);
  transpose_cast_kernel<<<dim3(LC_ / 32, D_ / 32, B_), 256, 0, stream>>>(C, w4mlu, CtWb, LC_);
  transpose_cast_kernel<<<dim3(LQ_ / 32, D_ / 32, B_), 256, 0, stream>>>(Q, nullptr, Qtb, LQ_);

  // Pass 2: similarity matrix S (WMMA)
  s_gemm_kernel<<<(B_ * (LC_ / 16) * (LQ_ / 64)) / 8, 256, 0, stream>>>(
      CtWb, Qtb, s0, s1, bias, S);

  // Pass 3: softmaxes
  row_softmax_kernel<<<dim3(LC_ / 8, B_), 256, 0, stream>>>(S, Qmask, S1b);
  col_softmax_kernel<<<dim3(LQ_ / 64, B_), 64, 0, stream>>>(S, Cmask, S2t);

  // Pass 4: T = S2^T @ Ct (WMMA), stored transposed as Tt (d, q)
  t_gemm_kernel<<<(B_ * (LQ_ / 16) * (D_ / 64)) / 8, 256, 0, stream>>>(S2t, Cb, Tt);

  // Pass 5: A = S1 @ Qt and Bm = S1 @ T (same WMMA kernel, different B operand)
  a_gemm_kernel<<<(B_ * (LC_ / 16) * (D_ / 64)) / 8, 256, 0, stream>>>(S1b, Qb, A);
  a_gemm_kernel<<<(B_ * (LC_ / 16) * (D_ / 64)) / 8, 256, 0, stream>>>(S1b, Tt, Bm);

  // Pass 6: assemble output
  output_kernel<<<dim3(LC_ / 32, B_), 256, 0, stream>>>(C, A, Bm, out);
}
